// SelfAttention2_52690658787774
// MI455X (gfx1250) — matrix operand order repaired and checked by reading, not compiled, not run on hardware
//
#include <hip/hip_runtime.h>

#define B_ 16
#define N_ 4096
#define D_ 128

// LDS row strides (bf16 elements) padded for conflict-free ds_load_b128:
#define KS_STRIDE 136   // 128 + 8  -> 272B row: 4-bank rotation
#define VT_STRIDE 40    // 32 + 8   -> 80B row: 20-bank rotation (distinct mod 64)
#define P_STRIDE  40

typedef __attribute__((ext_vector_type(16))) __bf16 v16bf;
typedef __attribute__((ext_vector_type(8)))  float  v8f;
typedef __attribute__((ext_vector_type(4)))  float  f32x4;
typedef __attribute__((ext_vector_type(4)))  unsigned int u32x4;

// Load two contiguous 8x bf16 chunks (16B each) into the A/B operand register
// shape expected by v_wmma_f32_16x16x32_bf16.
__device__ __forceinline__ v16bf load16bf(const __bf16* p0, const __bf16* p1) {
  union { v16bf v; u32x4 q[2]; } u;
  u.q[0] = *(const u32x4*)p0;
  u.q[1] = *(const u32x4*)p1;
  return u.v;
}

// CDNA5 async global->LDS copy (16B per lane), tracked by ASYNCcnt.
__device__ __forceinline__ void async_load_b128(unsigned int lds_off,
                                                const void* gptr) {
  asm volatile("global_load_async_to_lds_b128 %0, %1, off"
               :: "v"(lds_off), "v"(gptr)
               : "memory");
}

__device__ __forceinline__ void wait_asynccnt0() {
  asm volatile("s_wait_asynccnt 0x0" ::: "memory");
}

// ---------------------------------------------------------------------------
// Phase 1: QKV projection.  One wave computes a 16x128 tile of Q, K or V.
// q = x @ W^T + b.  Q,K row-major bf16 [B*N, D]; V stored transposed [B, D, N].
// ---------------------------------------------------------------------------
__global__ __launch_bounds__(32) void qkv_proj_kernel(
    const float* __restrict__ X,
    const float* __restrict__ Wq, const float* __restrict__ bq,
    const float* __restrict__ Wk, const float* __restrict__ bk,
    const float* __restrict__ Wv, const float* __restrict__ bv,
    __bf16* __restrict__ Q, __bf16* __restrict__ K, __bf16* __restrict__ VT)
{
  const int r    = blockIdx.x;         // 16-row tile index over B*N
  const int m    = blockIdx.y;         // 0=Q 1=K 2=V
  const int lane = threadIdx.x;
  const int lr   = lane & 15;
  const int hi   = lane >> 4;
  const int n0   = r * 16;
  const int b    = n0 / N_;
  const int nl0  = n0 % N_;

  const float* W    = (m == 0) ? Wq : (m == 1) ? Wk : Wv;
  const float* bias = (m == 0) ? bq : (m == 1) ? bk : bv;

  // A operand tiles: rows = lane%16, contraction k = kk*32 + 16*(lane/16) + i (contiguous, same as B).
  v16bf a[4];
  const float* xrow = X + (size_t)(n0 + lr) * D_;
#pragma unroll
  for (int kk = 0; kk < 4; ++kk) {
    const int c1 = kk * 32 + hi * 16;
    union { v16bf v; __bf16 e[16]; } u;
    f32x4 f0 = *(const f32x4*)(xrow + c1);
    f32x4 f1 = *(const f32x4*)(xrow + c1 + 4);
    f32x4 f2 = *(const f32x4*)(xrow + c1 + 8);
    f32x4 f3 = *(const f32x4*)(xrow + c1 + 12);
#pragma unroll
    for (int i = 0; i < 4; ++i) {
      u.e[i]      = (__bf16)f0[i];
      u.e[4 + i]  = (__bf16)f1[i];
      u.e[8 + i]  = (__bf16)f2[i];
      u.e[12 + i] = (__bf16)f3[i];
    }
    a[kk] = u.v;
  }

#pragma unroll
  for (int dt = 0; dt < 8; ++dt) {
    const int dcol = dt * 16 + lr;      // output column for this lane (B/C N-dim)
    v8f c = {};
#pragma unroll
    for (int kk = 0; kk < 4; ++kk) {
      // B operand: element i -> contraction k = kk*32 + hi*16 + i, value W[dcol, k]
      const float* wrow = W + (size_t)dcol * D_ + kk * 32 + hi * 16;
      union { v16bf v; __bf16 e[16]; } u;
#pragma unroll
      for (int i = 0; i < 16; i += 4) {
        f32x4 f = *(const f32x4*)(wrow + i);
#pragma unroll
        for (int t = 0; t < 4; ++t) u.e[i + t] = (__bf16)f[t];
      }
      c = __builtin_amdgcn_wmma_f32_16x16x32_bf16(false, a[kk], false, u.v,
                                                  (short)0, c, false, false);
    }
    const float bv_ = bias[dcol];
#pragma unroll
    for (int j = 0; j < 8; ++j) {
      const float v  = c[j] + bv_;
      const int  row = j + 8 * hi;      // C layout: M = j + 8*(lane/16)
      if (m == 2) {
        VT[((size_t)b * D_ + dcol) * N_ + (nl0 + row)] = (__bf16)v;
      } else {
        __bf16* dst = (m == 0) ? Q : K;
        dst[(size_t)(n0 + row) * D_ + dcol] = (__bf16)v;
      }
    }
  }
}

// ---------------------------------------------------------------------------
// Phase 2: flash attention. 4 waves per block share double-buffered K/V tiles
// staged into LDS with async global->LDS copies. Each wave owns 16 query rows;
// online softmax over 32-key blocks; per-tile column sums written as fp32
// partials [B, N/16, D] for the final mean.
// ---------------------------------------------------------------------------
__global__ __launch_bounds__(128) void attn_kernel(
    const __bf16* __restrict__ Q, const __bf16* __restrict__ K,
    const __bf16* __restrict__ VT, float* __restrict__ partial)
{
  __shared__ __bf16 Ks[2][32 * KS_STRIDE];    // K block: [32 keys][128 d] padded
  __shared__ __bf16 Vs[2][D_ * VT_STRIDE];    // V block: [128 d][32 keys] padded
  __shared__ __bf16 Ps[4][16 * P_STRIDE];     // per-wave P bounce (C->A relayout)

  const int blk  = blockIdx.x;
  const int b    = blk >> 6;                  // 64 blocks per batch
  const int tid  = threadIdx.x;
  const int wave = tid >> 5;
  const int lane = tid & 31;
  const int lr   = lane & 15;
  const int hi   = lane >> 4;
  const int qt   = (blk & 63) * 4 + wave;     // query tile within batch
  const size_t g0 = (size_t)b * N_ + qt * 16;

  const __bf16* Kg = K  + (size_t)b * N_ * D_;
  const __bf16* Vg = VT + (size_t)b * D_ * N_;

  // Async stage of one 32-key block (K: 8KB, V: 8KB) into LDS buffer `buf`.
  auto stage = [&](int buf, int k0) {
    const unsigned int kbase = (unsigned int)(uintptr_t)&Ks[buf][0];
#pragma unroll
    for (int r = 0; r < 4; ++r) {
      const int c   = r * 128 + tid;          // 16B chunk id, 0..511
      const int row = c >> 4, part = c & 15;
      async_load_b128(kbase + (unsigned int)(row * KS_STRIDE + part * 8) * 2,
                      Kg + (size_t)(k0 + row) * D_ + part * 8);
    }
    const unsigned int vbase = (unsigned int)(uintptr_t)&Vs[buf][0];
#pragma unroll
    for (int r = 0; r < 4; ++r) {
      const int c = r * 128 + tid;            // 16B chunk id, 0..511
      const int d = c >> 2, part = c & 3;
      async_load_b128(vbase + (unsigned int)(d * VT_STRIDE + part * 8) * 2,
                      Vg + (size_t)d * N_ + k0 + part * 8);
    }
  };

  stage(0, 0);                                // prologue: fill buffer 0

  // Q as A-operand tiles (kept in registers for the whole key loop)
  v16bf qa[4];
  const __bf16* qrow = Q + (g0 + lr) * D_;
#pragma unroll
  for (int kk = 0; kk < 4; ++kk) {
    const int c1 = kk * 32 + hi * 16;
    qa[kk] = load16bf(qrow + c1, qrow + c1 + 8);
  }

  v8f ot[8];
#pragma unroll
  for (int t = 0; t < 8; ++t) ot[t] = (v8f){};
  float mrun[8], lrun[8];
#pragma unroll
  for (int j = 0; j < 8; ++j) { mrun[j] = -1e30f; lrun[j] = 0.0f; }

  const float LOG2E = 1.44269504f;
  __bf16* Pw = &Ps[wave][0];

  for (int kb = 0; kb < N_ / 32; ++kb) {
    const int cur = kb & 1;

    wait_asynccnt0();      // this wave's stage(kb) has landed in LDS
    __syncthreads();       // all waves staged; all waves done reading buf cur^1
    if (kb + 1 < N_ / 32)
      stage(cur ^ 1, (kb + 1) * 32);          // overlap next transfer w/ compute

    // ---- S = Q @ K^T for 32 keys (two 16x16 C tiles), K from LDS ----
    v8f s0 = {}, s1 = {};
    const __bf16* ks0 = &Ks[cur][lr * KS_STRIDE + hi * 16];
    const __bf16* ks1 = &Ks[cur][(16 + lr) * KS_STRIDE + hi * 16];
#pragma unroll
    for (int kk = 0; kk < 4; ++kk) {
      v16bf kt0 = load16bf(ks0 + kk * 32, ks0 + kk * 32 + 8);
      s0 = __builtin_amdgcn_wmma_f32_16x16x32_bf16(false, qa[kk], false, kt0,
                                                   (short)0, s0, false, false);
      v16bf kt1 = load16bf(ks1 + kk * 32, ks1 + kk * 32 + 8);
      s1 = __builtin_amdgcn_wmma_f32_16x16x32_bf16(false, qa[kk], false, kt1,
                                                   (short)0, s1, false, false);
    }

    // ---- online softmax over the 32-key block ----
    float alpha[8];
#pragma unroll
    for (int j = 0; j < 8; ++j) {
      float mx = fmaxf(s0[j], s1[j]);
#pragma unroll
      for (int off = 1; off < 16; off <<= 1)
        mx = fmaxf(mx, __shfl_xor(mx, off, 32));
      const float mnew = fmaxf(mrun[j], mx);
      alpha[j] = exp2f((mrun[j] - mnew) * LOG2E);
      const float p0 = exp2f((s0[j] - mnew) * LOG2E);
      const float p1 = exp2f((s1[j] - mnew) * LOG2E);
      Pw[(j + 8 * hi) * P_STRIDE + lr]      = (__bf16)p0;
      Pw[(j + 8 * hi) * P_STRIDE + 16 + lr] = (__bf16)p1;
      float rsum = p0 + p1;
#pragma unroll
      for (int off = 1; off < 16; off <<= 1)
        rsum += __shfl_xor(rsum, off, 32);
      lrun[j] = lrun[j] * alpha[j] + rsum;
      mrun[j] = mnew;
    }
#pragma unroll
    for (int t = 0; t < 8; ++t)
#pragma unroll
      for (int j = 0; j < 8; ++j) ot[t][j] *= alpha[j];

    // P reloaded in A-operand layout (intra-wave DS ops are in-order; the
    // per-wave P buffer needs no workgroup barrier)
    v16bf pa = load16bf(&Pw[lr * P_STRIDE + hi * 16],
                        &Pw[lr * P_STRIDE + hi * 16 + 8]);

    // ---- O += P @ V  (V transposed in LDS: contraction over keys contiguous) --
    const __bf16* vs = &Vs[cur][hi * 16];
#pragma unroll
    for (int t = 0; t < 8; ++t) {
      const __bf16* vp = vs + (t * 16 + lr) * VT_STRIDE;
      v16bf vt = load16bf(vp, vp + 8);
      ot[t] = __builtin_amdgcn_wmma_f32_16x16x32_bf16(false, pa, false, vt,
                                                      (short)0, ot[t], false, false);
    }
  }

  // ---- normalize rows and reduce the 16 rows (for the final mean over N) ----
  float inv[8];
#pragma unroll
  for (int j = 0; j < 8; ++j) inv[j] = 1.0f / lrun[j];
#pragma unroll
  for (int t = 0; t < 8; ++t) {
    float cs = 0.0f;
#pragma unroll
    for (int j = 0; j < 8; ++j) cs += ot[t][j] * inv[j];
    cs += __shfl_xor(cs, 16, 32);      // add the other 8 rows (other half-wave)
    if (hi == 0)
      partial[((size_t)(b * (N_ / 16) + qt)) * D_ + t * 16 + lr] = cs;
  }
}

// ---------------------------------------------------------------------------
// Phase 3: deterministic reduction of per-tile partials -> out[b, d] (mean).
// ---------------------------------------------------------------------------
__global__ __launch_bounds__(256) void reduce_kernel(
    const float* __restrict__ partial, float* __restrict__ out)
{
  const int idx = blockIdx.x * blockDim.x + threadIdx.x;   // 0 .. B*D-1
  const int b = idx / D_;
  const int d = idx % D_;
  float s = 0.0f;
  for (int qtile = 0; qtile < N_ / 16; ++qtile)
    s += partial[((size_t)(b * (N_ / 16) + qtile)) * D_ + d];
  out[idx] = s * (1.0f / (float)N_);
}

extern "C" void kernel_launch(void* const* d_in, const int* in_sizes, int n_in,
                              void* d_out, int out_size, void* d_ws, size_t ws_size,
                              hipStream_t stream) {
  const float* X  = (const float*)d_in[0];
  const float* Wq = (const float*)d_in[1];
  const float* bq = (const float*)d_in[2];
  const float* Wk = (const float*)d_in[3];
  const float* bk = (const float*)d_in[4];
  const float* Wv = (const float*)d_in[5];
  const float* bv = (const float*)d_in[6];

  char* ws = (char*)d_ws;
  const size_t qkv_bytes = (size_t)B_ * N_ * D_ * 2;   // 16 MB each, bf16
  __bf16* Q  = (__bf16*)(ws);
  __bf16* K  = (__bf16*)(ws + qkv_bytes);
  __bf16* VT = (__bf16*)(ws + 2 * qkv_bytes);
  float* partial = (float*)(ws + 3 * qkv_bytes);       // B * (N/16) * D fp32

  dim3 g1(B_ * N_ / 16, 3);
  qkv_proj_kernel<<<g1, 32, 0, stream>>>(X, Wq, bq, Wk, bk, Wv, bv, Q, K, VT);
  attn_kernel<<<(B_ * (N_ / 16)) / 4, 128, 0, stream>>>(Q, K, VT, partial);
  reduce_kernel<<<(B_ * D_) / 256, 256, 0, stream>>>(partial, (float*)d_out);
}
